// SpikingSiameseNetwork_73916387164331
// MI455X (gfx1250) — matrix operand order repaired and checked by reading, not compiled, not run on hardware
//
#include <hip/hip_runtime.h>

typedef __attribute__((ext_vector_type(16))) _Float16 v16h;
typedef __attribute__((ext_vector_type(8)))  _Float16 v8h;
typedef __attribute__((ext_vector_type(4)))  _Float16 v4h;
typedef __attribute__((ext_vector_type(8)))  float    v8f;
typedef __attribute__((ext_vector_type(4)))  float    v4f;
typedef __attribute__((ext_vector_type(4)))  unsigned int u32x4;
typedef __attribute__((ext_vector_type(8)))  int      i32x8;
typedef __attribute__((ext_vector_type(4)))  int      i32x4;

#define LIF_BETA   0.9f
#define LIF_THRESH 1.0f

static __device__ __forceinline__ v8f wmma16(v16h a, v16h b, v8f c) {
    // v_wmma_f32_16x16x32_f16 : D = A(16x32) * B(32x16) + C(16x16 f32)
    return __builtin_amdgcn_wmma_f32_16x16x32_f16(false, a, false, b, (short)0, c, false, false);
}

static __device__ __forceinline__ v8f splat8(float x) {
    v8f v = {x, x, x, x, x, x, x, x};
    return v;
}

// 16-bit A-matrix fragment (16x32 MxK), contiguous k = 16*(lane/16) + s.
static __device__ __forceinline__ v16h load_a_frag(const _Float16* p) {
    v8h lo = *(const v8h*)(p);
    v8h hi = *(const v8h*)(p + 8);
    return __builtin_shufflevector(lo, hi, 0,1,2,3,4,5,6,7,8,9,10,11,12,13,14,15);
}

// TDM: async DMA of a 2D tile (rows x 512 f16, row stride 25600B) into LDS.
// D# built per CDNA5 ISA ch.8: group0 = {flags, lds_addr, global_addr, type},
// group1 = {data_size, tensor dims/strides, tile dims}. Issued wave-uniform.
static __device__ __forceinline__ void tdm_load_tile(unsigned ldsOff, const _Float16* gptr) {
    unsigned long long ga = (unsigned long long)(const void*)gptr;
    u32x4 g0 = {1u,                                   // count=1, user descriptor
                ldsOff,                               // lds_addr (bytes)
                (unsigned)(ga & 0xFFFFFFFFu),         // global_addr[31:0]
                (((unsigned)(ga >> 32)) & 0x01FFFFFFu) | 0x80000000u}; // addr[56:32] | type=2
    i32x8 g1 = {(int)(1u << 16),        // data_size = 2 bytes
                (int)(512u << 16),      // tensor_dim0[15:0] (=512) at bits 79:48
                (int)(16u << 16),       // tensor_dim1[15:0] (=16) at bits 111:80
                (int)(512u << 16),      // tile_dim0 = 512 at bits 127:112
                16,                     // tile_dim1 = 16, tile_dim2 = 0
                12800,                  // tensor_dim0_stride = 25600B / 2
                0, 0};                  // dim1_stride = 0
    i32x4 z4 = {0, 0, 0, 0};
#if defined(__clang_major__) && (__clang_major__ >= 23)
    i32x8 z8 = {0, 0, 0, 0, 0, 0, 0, 0};
    __builtin_amdgcn_tensor_load_to_lds(g0, g1, z4, z4, z8, 0);
#else
    __builtin_amdgcn_tensor_load_to_lds(g0, g1, z4, z4, 0);
#endif
}

// Pack f32 weight [OUT][IN] into WMMA-B-fragment-contiguous f16:
// dst[(cb*(IN/32)+ks)*512 + lane*16 + j] = W[cb*16 + lane%16][ks*32 + (lane/16)*16 + j]
__global__ void pack_w_kernel(const float* __restrict__ W, _Float16* __restrict__ dst,
                              int OUT, int IN) {
    int d = blockIdx.x * blockDim.x + threadIdx.x;
    if (d >= OUT * IN) return;
    int j    = d & 15;
    int l    = (d >> 4) & 31;
    int rest = d >> 9;
    int ksteps = IN >> 5;
    int ks = rest % ksteps;
    int cb = rest / ksteps;
    int n = cb * 16 + (l & 15);
    int k = ks * 32 + (l >> 4) * 16 + j;
    dst[d] = (_Float16)W[(size_t)n * IN + k];
}

// One-time f32 -> f16 conversion of the spike input streams (values are 0/1).
__global__ void cvt_f16_kernel(const float* __restrict__ src, _Float16* __restrict__ dst,
                               long n4) {
    long i = (long)blockIdx.x * blockDim.x + threadIdx.x;
    if (i >= n4) return;
    float4 f = ((const float4*)src)[i];
    v4h h = {(_Float16)f.x, (_Float16)f.y, (_Float16)f.z, (_Float16)f.w};
    ((v4h*)dst)[i] = h;
}

// LIF on one 16x16 tile held as C-fragment (v8f). Membrane lives in LDS in
// fragment order: memLds[(cb*32 + lane)*8 + e] (32B contiguous per lane).
static __device__ __forceinline__ void lif_tile_spk(
    v8f c, float* __restrict__ memLds, _Float16* __restrict__ sSpk,
    int cb, int lane, int hlf, int n16)
{
    float* mf = memLds + ((size_t)(cb * 32 + lane)) * 8;
    v4f lo = *(const v4f*)mf;
    v4f hi = *(const v4f*)(mf + 4);
    v8f m = __builtin_shufflevector(lo, hi, 0,1,2,3,4,5,6,7);
#pragma unroll
    for (int e = 0; e < 8; ++e) {
        float mo  = m[e];
        float rst = (mo > LIF_THRESH) ? LIF_THRESH : 0.0f;
        float mn  = LIF_BETA * mo + c[e] - rst;
        m[e] = mn;
        sSpk[(e + hlf * 8) * 256 + cb * 16 + n16] =
            (mn > LIF_THRESH) ? (_Float16)1.0f : (_Float16)0.0f;
    }
    *(v4f*)mf       = __builtin_shufflevector(m, m, 0,1,2,3);
    *(v4f*)(mf + 4) = __builtin_shufflevector(m, m, 4,5,6,7);
}

static __device__ __forceinline__ void lif_tile_fin(
    v8f c, float* __restrict__ memLds, float* __restrict__ sFin, int colOff,
    bool writeFin, int cb, int lane, int hlf, int n16)
{
    float* mf = memLds + ((size_t)(cb * 32 + lane)) * 8;
    v4f lo = *(const v4f*)mf;
    v4f hi = *(const v4f*)(mf + 4);
    v8f m = __builtin_shufflevector(lo, hi, 0,1,2,3,4,5,6,7);
#pragma unroll
    for (int e = 0; e < 8; ++e) {
        float mo  = m[e];
        float rst = (mo > LIF_THRESH) ? LIF_THRESH : 0.0f;
        float mn  = LIF_BETA * mo + c[e] - rst;
        m[e] = mn;
        if (writeFin) {
            sFin[(e + hlf * 8) * 256 + colOff + cb * 16 + n16] =
                (mn > LIF_THRESH) ? 1.0f : 0.0f;
        }
    }
    *(v4f*)mf       = __builtin_shufflevector(m, m, 0,1,2,3);
    *(v4f*)(mf + 4) = __builtin_shufflevector(m, m, 4,5,6,7);
}

// Layer1: 16 rows, K=512 -> H=256; wave owns 4 col-blocks, processed in pairs.
static __device__ __forceinline__ void layer1_pass(
    const _Float16* __restrict__ sIn, const _Float16* __restrict__ Wp,
    _Float16* __restrict__ sSpk, const float* __restrict__ bias,
    float* __restrict__ memLds, int wave, int lane, int hlf, int n16)
{
#pragma unroll
    for (int jp = 0; jp < 2; ++jp) {
        const int cb0 = wave * 4 + jp * 2;
        const int cb1 = cb0 + 1;
        v8f c0 = splat8(bias[cb0 * 16 + n16]);
        v8f c1 = splat8(bias[cb1 * 16 + n16]);
#pragma unroll 4
        for (int ks = 0; ks < 16; ++ks) {
            v16h af = load_a_frag(sIn + n16 * 512 + ks * 32 + hlf * 16);
            v16h b0 = *(const v16h*)(Wp + (size_t)(cb0 * 16 + ks) * 512 + lane * 16);
            v16h b1 = *(const v16h*)(Wp + (size_t)(cb1 * 16 + ks) * 512 + lane * 16);
            c0 = wmma16(af, b0, c0);
            c1 = wmma16(af, b1, c1);
        }
        lif_tile_spk(c0, memLds, sSpk, cb0, lane, hlf, n16);
        lif_tile_spk(c1, memLds, sSpk, cb1, lane, hlf, n16);
    }
}

// Layer2: 16 rows, K=256 -> E=128; wave owns 2 col-blocks (one pair).
static __device__ __forceinline__ void layer2_pass(
    const _Float16* __restrict__ sSpk, const _Float16* __restrict__ Wp,
    const float* __restrict__ bias, float* __restrict__ memLds,
    float* __restrict__ sFin, int colOff, bool writeFin,
    int wave, int lane, int hlf, int n16)
{
    const int cb0 = wave * 2;
    const int cb1 = cb0 + 1;
    v8f c0 = splat8(bias[cb0 * 16 + n16]);
    v8f c1 = splat8(bias[cb1 * 16 + n16]);
#pragma unroll 4
    for (int ks = 0; ks < 8; ++ks) {
        v16h af = load_a_frag(sSpk + n16 * 256 + ks * 32 + hlf * 16);
        v16h b0 = *(const v16h*)(Wp + (size_t)(cb0 * 8 + ks) * 512 + lane * 16);
        v16h b1 = *(const v16h*)(Wp + (size_t)(cb1 * 8 + ks) * 512 + lane * 16);
        c0 = wmma16(af, b0, c0);
        c1 = wmma16(af, b1, c1);
    }
    lif_tile_fin(c0, memLds, sFin, colOff, writeFin, cb0, lane, hlf, n16);
    lif_tile_fin(c1, memLds, sFin, colOff, writeFin, cb1, lane, hlf, n16);
}

__global__ __launch_bounds__(128) void snn_fused_kernel(
    const _Float16* __restrict__ a16, const _Float16* __restrict__ v16,
    const float* __restrict__ b_a1, const float* __restrict__ b_a2,
    const float* __restrict__ b_v1, const float* __restrict__ b_v2,
    const _Float16* __restrict__ Wa1p, const _Float16* __restrict__ Wv1p,
    const _Float16* __restrict__ Wa2p, const _Float16* __restrict__ Wv2p,
    float* __restrict__ out)
{
    constexpr int T = 25, A = 512;

    // double-buffered TDM destination: [stream][buf][16 rows][512]
    __shared__ __align__(16) _Float16 sIn[2][2][16 * 512];   // 64 KB
    __shared__ __align__(16) _Float16 sSpkA[16 * 256];       //  8 KB
    __shared__ __align__(16) _Float16 sSpkV[16 * 256];       //  8 KB
    __shared__ __align__(16) float sFin[16 * 256];           // 16 KB
    // membranes in C-fragment order [cb][lane][8]
    __shared__ __align__(16) float memA1[16 * 32 * 8];       // 16 KB
    __shared__ __align__(16) float memV1[16 * 32 * 8];       // 16 KB
    __shared__ __align__(16) float memA2[8 * 32 * 8];        //  8 KB
    __shared__ __align__(16) float memV2[8 * 32 * 8];        //  8 KB
    __shared__ float sPart[16 * 8];
    __shared__ float sInv[16];

    const int tid  = threadIdx.x;
    const int wave = tid >> 5;
    const int lane = tid & 31;
    const int hlf  = lane >> 4;
    const int n16  = lane & 15;
    const long row0 = (long)blockIdx.x * 16;   // 128 WGs x 16 rows = BS 2048

    // zero membranes
    for (int i = tid; i < 16 * 32 * 8; i += 128) { memA1[i] = 0.0f; memV1[i] = 0.0f; }
    for (int i = tid; i < 8 * 32 * 8;  i += 128) { memA2[i] = 0.0f; memV2[i] = 0.0f; }

    // TDM prologue: kick off DMA for timestep 0 into buffer 0 (wave 0 only)
    const long tileBase = row0 * T * (long)A;   // halfs
    if (wave == 0) {
        tdm_load_tile((unsigned)(size_t)&sIn[0][0][0], a16 + tileBase);
        tdm_load_tile((unsigned)(size_t)&sIn[1][0][0], v16 + tileBase);
    }

    for (int t = 0; t < T; ++t) {
        const int buf = t & 1;
        if (wave == 0) __builtin_amdgcn_s_wait_tensorcnt((unsigned short)0);
        __syncthreads();   // DMA'd inputs visible; prev iter's LDS reads done

        // overlap: DMA next timestep into the other buffer while we compute
        if (wave == 0 && (t + 1) < T) {
            tdm_load_tile((unsigned)(size_t)&sIn[0][buf ^ 1][0],
                          a16 + tileBase + (size_t)(t + 1) * A);
            tdm_load_tile((unsigned)(size_t)&sIn[1][buf ^ 1][0],
                          v16 + tileBase + (size_t)(t + 1) * A);
        }

        // ---- layer 1 (WMMA over K=512) + LIF + spikes to LDS ----
        layer1_pass(sIn[0][buf], Wa1p, sSpkA, b_a1, memA1, wave, lane, hlf, n16);
        layer1_pass(sIn[1][buf], Wv1p, sSpkV, b_v1, memV1, wave, lane, hlf, n16);
        __syncthreads();

        // ---- layer 2 (WMMA over K=256) + LIF ----
        const bool last = (t == T - 1);
        layer2_pass(sSpkA, Wa2p, b_a2, memA2, sFin, 0,   last, wave, lane, hlf, n16);
        layer2_pass(sSpkV, Wv2p, b_v2, memV2, sFin, 128, last, wave, lane, hlf, n16);
    }
    __syncthreads();

    // ---- L2-normalize concat spikes per row, then mean over S=8 ----
    {
        int r = tid >> 3, seg = tid & 7;
        float s = 0.0f;
#pragma unroll
        for (int c = 0; c < 32; ++c) s += sFin[r * 256 + seg * 32 + c];
        sPart[r * 8 + seg] = s;
    }
    __syncthreads();
    if (tid < 16) {
        float s = 0.0f;
#pragma unroll
        for (int i = 0; i < 8; ++i) s += sPart[tid * 8 + i];
        sInv[tid] = 1.0f / fmaxf(sqrtf(s), 1e-12f);
    }
    __syncthreads();
#pragma unroll
    for (int q = 0; q < 4; ++q) {
        int o = tid + q * 128;          // 512 outputs per WG (2 b-rows x 256)
        int bl = o >> 8, c = o & 255;
        float acc = 0.0f;
#pragma unroll
        for (int s2 = 0; s2 < 8; ++s2)
            acc += sFin[(bl * 8 + s2) * 256 + c] * sInv[bl * 8 + s2];
        out[((row0 >> 3) + bl) * 256 + c] = acc * 0.125f;
    }
}

extern "C" void kernel_launch(void* const* d_in, const int* in_sizes, int n_in,
                              void* d_out, int out_size, void* d_ws, size_t ws_size,
                              hipStream_t stream) {
    const float* a_seq = (const float*)d_in[0];
    const float* v_seq = (const float*)d_in[1];
    const float* W_a1  = (const float*)d_in[2];
    const float* b_a1  = (const float*)d_in[3];
    const float* W_a2  = (const float*)d_in[4];
    const float* b_a2  = (const float*)d_in[5];
    const float* W_v1  = (const float*)d_in[6];
    const float* b_v1  = (const float*)d_in[7];
    const float* W_v2  = (const float*)d_in[8];
    const float* b_v2  = (const float*)d_in[9];
    float* out = (float*)d_out;

    // scratch layout (halfs): packed weights (640KB) then f16 input streams
    const long N_IN = 256L * 8 * 25 * 512;      // 26,214,400 elements per stream
    _Float16* ws   = (_Float16*)d_ws;
    _Float16* Wa1p = ws;                  // 256*512
    _Float16* Wv1p = ws + 131072;         // 256*512
    _Float16* Wa2p = ws + 262144;         // 128*256
    _Float16* Wv2p = ws + 294912;         // 128*256
    _Float16* a16  = ws + 327680;         // N_IN halfs
    _Float16* v16  = a16 + N_IN;          // N_IN halfs

    pack_w_kernel<<<512, 256, 0, stream>>>(W_a1, Wa1p, 256, 512);
    pack_w_kernel<<<512, 256, 0, stream>>>(W_v1, Wv1p, 256, 512);
    pack_w_kernel<<<128, 256, 0, stream>>>(W_a2, Wa2p, 128, 256);
    pack_w_kernel<<<128, 256, 0, stream>>>(W_v2, Wv2p, 128, 256);

    const long n4 = N_IN / 4;
    cvt_f16_kernel<<<(unsigned)((n4 + 255) / 256), 256, 0, stream>>>(a_seq, a16, n4);
    cvt_f16_kernel<<<(unsigned)((n4 + 255) / 256), 256, 0, stream>>>(v_seq, v16, n4);

    snn_fused_kernel<<<128, 128, 0, stream>>>(
        a16, v16, b_a1, b_a2, b_v1, b_v2, Wa1p, Wv1p, Wa2p, Wv2p, out);
}
